// Block_89008902242484
// MI455X (gfx1250) — matrix operand order repaired and checked by reading, not compiled, not run on hardware
//
#include <hip/hip_runtime.h>
#include <hip/hip_bf16.h>
#include <math.h>
#include <stdint.h>

typedef __attribute__((ext_vector_type(16))) _Float16 v16h;
typedef __attribute__((ext_vector_type(8)))  _Float16 v8h;
typedef __attribute__((ext_vector_type(8)))  float    v8f;
typedef __attribute__((ext_vector_type(4)))  float    v4f;
typedef __attribute__((__vector_size__(16))) int      vi4;

// ---------------------------------------------------------------- constants
constexpr int kB  = 8;
constexpr int kC  = 256;
constexpr int kH  = 32;
constexpr int kW  = 32;
constexpr int kHW = kH * kW;          // 1024
constexpr int kL  = kHW;              // sequence length
constexpr int kNT = kB * kHW;         // 8192 tokens
constexpr int kDI = 512;              // d_inner
constexpr int kDS = 16;               // d_state (== DT_RANK)
constexpr int kLdXdbl = 48;           // DT_RANK + 2*D_STATE
constexpr int kMlpH = 1024;

#define ACT_NONE 0
#define ACT_GELU 1
#define ACT_SOFTPLUS 2
#define ACT_SILU 3

#if __has_builtin(__builtin_amdgcn_global_load_async_to_lds_b128) && \
    __has_builtin(__builtin_amdgcn_s_wait_asynccnt)
#define USE_ASYNC_LDS 1
#else
#define USE_ASYNC_LDS 0
#endif

__device__ __forceinline__ float act_apply(float v, int act) {
  switch (act) {
    case ACT_GELU:     return 0.5f * v * (1.f + erff(v * 0.7071067811865475f));
    case ACT_SOFTPLUS: return (v > 20.f) ? v : log1pf(__expf(v));
    case ACT_SILU:     return v * (1.f / (1.f + __expf(-v)));
    default:           return v;
  }
}

// async copy of one 16B chunk of a weight tile: global(f16) -> LDS
__device__ __forceinline__ void w_tile_load(const _Float16* g, _Float16* l) {
#if USE_ASYNC_LDS
  __builtin_amdgcn_global_load_async_to_lds_b128(
      (__attribute__((address_space(1))) vi4*)(uintptr_t)g,
      (__attribute__((address_space(3))) vi4*)(uint32_t)(uintptr_t)l, 0, 0);
#else
  *(v8h*)l = *(const v8h*)g;
#endif
}
__device__ __forceinline__ void w_tile_wait() {
#if USE_ASYNC_LDS
  __builtin_amdgcn_s_wait_asynccnt(0);
#endif
}

// ------------------------------------------------------------ WMMA GEMM
// Y[M,N] = act(A[M,K](f32,lda) * W[N,K](f16,ldw)^T + bias), f32 out.
// 256 threads = 8 waves; block tile 64x64, BK=32; wave tile 16x32 (2 WMMAs).
// Fast path: A via reg pipeline (b128 loads + cvt), W via async LDS copy with
// double-buffered Bs. Generic path: per-thread-uniform guards -> b128 loads.
__global__ void __launch_bounds__(256) k_gemm_wmma(
    const float* __restrict__ A, int lda,
    const _Float16* __restrict__ Wt, int ldw,
    const float* __restrict__ bias,
    float* __restrict__ Y, int ldy,
    int M, int N, int K, int act) {
  __shared__ __align__(16) _Float16 As[64][40];
  __shared__ __align__(16) _Float16 Bs[2][64][40];

  const int t    = threadIdx.x;
  const int m0   = blockIdx.y * 64;
  const int n0   = blockIdx.x * 64;
  const int lane = t & 31;
  const int wv   = t >> 5;
  const int wr   = wv & 3;   // 4 row tiles of 16
  const int wc   = wv >> 2;  // 2 col tiles of 32
  const int half = lane >> 4;
  const int r    = lane & 15;
  const int srow = t >> 2;        // staging row 0..63 (4 threads per row)
  const int sk   = (t & 3) * 8;   // staging k start (16B aligned in f16)

  v8f c0 = {0.f, 0.f, 0.f, 0.f, 0.f, 0.f, 0.f, 0.f};
  v8f c1 = {0.f, 0.f, 0.f, 0.f, 0.f, 0.f, 0.f, 0.f};

  const int KT = (K + 31) >> 5;
  const int gm = m0 + srow;
  const int gn = n0 + srow;
  const float* aRow = A + (size_t)gm * lda;
  const _Float16* wRow = Wt + (size_t)gn * ldw;
  _Float16* aSlot  = &As[srow][sk];
  _Float16* wSlot0 = &Bs[0][srow][sk];
  _Float16* wSlot1 = &Bs[1][srow][sk];

  const bool all_full = ((K & 31) == 0) && (m0 + 64 <= M) && (n0 + 64 <= N);

  if (all_full) {
    w_tile_load(wRow + sk, wSlot0);  // tile 0 -> Bs[0]
    v4f ra0 = *(const v4f*)(aRow + sk);
    v4f ra1 = *(const v4f*)(aRow + sk + 4);
    for (int kt = 0; kt < KT; ++kt) {
      const int cur = kt & 1;
      v8h ha;
#pragma unroll
      for (int i = 0; i < 4; ++i) {
        ha[i]     = (_Float16)ra0[i];
        ha[4 + i] = (_Float16)ra1[i];
      }
      *(v8h*)aSlot = ha;  // ds_store_b128
      w_tile_wait();      // tile kt's async W copy complete
      __syncthreads();
      if (kt + 1 < KT) {  // overlap next tile's loads with this tile's WMMAs
        const int kn = (kt + 1) * 32;
        w_tile_load(wRow + kn + sk, cur ? wSlot0 : wSlot1);
        ra0 = *(const v4f*)(aRow + kn + sk);
        ra1 = *(const v4f*)(aRow + kn + sk + 4);
        __builtin_prefetch(aRow + kn + 32 + sk, 0, 1);
      }
      // A frag 16x32: elem e -> K=16h+e
      // B frag 32x16: elem e -> K=16h+e
      const _Float16* ap  = &As[wr * 16 + r][0];
      const _Float16* bp0 = &Bs[cur][wc * 32 + r][0];
      const _Float16* bp1 = &Bs[cur][wc * 32 + 16 + r][0];
      v8h alo  = *(const v8h*)(ap + 16 * half);
      v8h ahi  = *(const v8h*)(ap + 16 * half + 8);
      v8h b0lo = *(const v8h*)(bp0 + 16 * half);
      v8h b0hi = *(const v8h*)(bp0 + 16 * half + 8);
      v8h b1lo = *(const v8h*)(bp1 + 16 * half);
      v8h b1hi = *(const v8h*)(bp1 + 16 * half + 8);
      v16h af  = __builtin_shufflevector(alo, ahi, 0, 1, 2, 3, 4, 5, 6, 7, 8,
                                         9, 10, 11, 12, 13, 14, 15);
      v16h bf0 = __builtin_shufflevector(b0lo, b0hi, 0, 1, 2, 3, 4, 5, 6, 7, 8,
                                         9, 10, 11, 12, 13, 14, 15);
      v16h bf1 = __builtin_shufflevector(b1lo, b1hi, 0, 1, 2, 3, 4, 5, 6, 7, 8,
                                         9, 10, 11, 12, 13, 14, 15);
      c0 = __builtin_amdgcn_wmma_f32_16x16x32_f16(false, af, false, bf0,
                                                  (short)0, c0, false, false);
      c1 = __builtin_amdgcn_wmma_f32_16x16x32_f16(false, af, false, bf1,
                                                  (short)0, c1, false, false);
      __syncthreads();
    }
  } else {
    for (int kt = 0; kt < KT; ++kt) {
      const int k0 = kt * 32;
      const int kRem = K - k0;
      // per-thread-uniform predicates -> vector loads in-branch, no exec dance
      if (gm < M && sk + 8 <= kRem) {
        v4f a0 = *(const v4f*)(aRow + k0 + sk);
        v4f a1 = *(const v4f*)(aRow + k0 + sk + 4);
        v8h ha;
#pragma unroll
        for (int i = 0; i < 4; ++i) {
          ha[i]     = (_Float16)a0[i];
          ha[4 + i] = (_Float16)a1[i];
        }
        *(v8h*)aSlot = ha;
      } else {
#pragma unroll
        for (int i = 0; i < 8; ++i) {
          const int gk = k0 + sk + i;
          aSlot[i] = (gm < M && gk < K) ? (_Float16)aRow[gk] : (_Float16)0.f;
        }
      }
      if (gn < N && sk + 8 <= kRem) {
        *(v8h*)wSlot0 = *(const v8h*)(wRow + k0 + sk);
      } else {
#pragma unroll
        for (int i = 0; i < 8; ++i) {
          const int gk = k0 + sk + i;
          wSlot0[i] = (gn < N && gk < K) ? wRow[gk] : (_Float16)0.f;
        }
      }
      __syncthreads();
      const _Float16* ap  = &As[wr * 16 + r][0];
      const _Float16* bp0 = &Bs[0][wc * 32 + r][0];
      const _Float16* bp1 = &Bs[0][wc * 32 + 16 + r][0];
      v8h alo  = *(const v8h*)(ap + 16 * half);
      v8h ahi  = *(const v8h*)(ap + 16 * half + 8);
      v8h b0lo = *(const v8h*)(bp0 + 16 * half);
      v8h b0hi = *(const v8h*)(bp0 + 16 * half + 8);
      v8h b1lo = *(const v8h*)(bp1 + 16 * half);
      v8h b1hi = *(const v8h*)(bp1 + 16 * half + 8);
      v16h af  = __builtin_shufflevector(alo, ahi, 0, 1, 2, 3, 4, 5, 6, 7, 8,
                                         9, 10, 11, 12, 13, 14, 15);
      v16h bf0 = __builtin_shufflevector(b0lo, b0hi, 0, 1, 2, 3, 4, 5, 6, 7, 8,
                                         9, 10, 11, 12, 13, 14, 15);
      v16h bf1 = __builtin_shufflevector(b1lo, b1hi, 0, 1, 2, 3, 4, 5, 6, 7, 8,
                                         9, 10, 11, 12, 13, 14, 15);
      c0 = __builtin_amdgcn_wmma_f32_16x16x32_f16(false, af, false, bf0,
                                                  (short)0, c0, false, false);
      c1 = __builtin_amdgcn_wmma_f32_16x16x32_f16(false, af, false, bf1,
                                                  (short)0, c1, false, false);
      __syncthreads();
    }
  }

  // C/D layout: vgpr i, lane(r,half) -> row = i + 8*half, col = r
  const int gn0 = n0 + wc * 32 + r;
  const int gn1 = gn0 + 16;
  const float b0 = (bias && gn0 < N) ? bias[gn0] : 0.f;
  const float b1 = (bias && gn1 < N) ? bias[gn1] : 0.f;
#pragma unroll
  for (int i = 0; i < 8; ++i) {
    const int gmi = m0 + wr * 16 + half * 8 + i;
    if (gmi < M) {
      if (gn0 < N) Y[(size_t)gmi * ldy + gn0] = act_apply(c0[i] + b0, act);
      if (gn1 < N) Y[(size_t)gmi * ldy + gn1] = act_apply(c1[i] + b1, act);
    }
  }
}

// ------------------------------------------------------- fp32 -> f16 convert
__global__ void k_cvt_h(const float* __restrict__ x, _Float16* __restrict__ y,
                        int n) {
  const int i = blockIdx.x * 256 + threadIdx.x;
  if (i < n) y[i] = (_Float16)x[i];
}

// ------------------------------------------------------- batchnorm (train)
__global__ void __launch_bounds__(256) k_bn(const float* __restrict__ x,
                                            const float* __restrict__ g,
                                            const float* __restrict__ be,
                                            float* __restrict__ y) {
  const int c = blockIdx.x;
  const int tid = threadIdx.x;
  __shared__ float s1[256], s2[256];
  __shared__ float mv[2];
  float a1 = 0.f, a2 = 0.f;
  for (int i = tid; i < kB * kHW; i += 256) {
    const int b = i / kHW, p = i - b * kHW;
    const float v = x[((size_t)(b * kC + c)) * kHW + p];
    a1 += v;
    a2 += v * v;
  }
  s1[tid] = a1;
  s2[tid] = a2;
  __syncthreads();
  for (int off = 128; off > 0; off >>= 1) {
    if (tid < off) { s1[tid] += s1[tid + off]; s2[tid] += s2[tid + off]; }
    __syncthreads();
  }
  if (tid == 0) {
    const float n = (float)(kB * kHW);
    const float m = s1[0] / n;
    mv[0] = m;
    mv[1] = rsqrtf(s2[0] / n - m * m + 1e-5f);
  }
  __syncthreads();
  const float m = mv[0], inv = mv[1], gg = g[c], bb = be[c];
  for (int i = tid; i < kB * kHW; i += 256) {
    const int b = i / kHW, p = i - b * kHW;
    const size_t a = ((size_t)(b * kC + c)) * kHW + p;
    y[a] = (x[a] - m) * inv * gg + bb;
  }
}

// -------------------------------------------------- layout transforms
__global__ void k_nchw2tok(const float* __restrict__ x, float* __restrict__ y,
                           int C, int mode) {
  const int i = blockIdx.x * 256 + threadIdx.x;
  const int total = kB * C * kHW;
  if (i >= total) return;
  const int p = i % kHW;
  const int c = (i / kHW) % C;
  const int b = i / (kHW * C);
  const int p2 = mode ? ((p % kW) * kH + (p / kW)) : p;  // mode1: (w,h) order
  y[((size_t)b * kHW + p2) * C + c] = x[i];
}

__global__ void k_tok2nchw(const float* __restrict__ t, float* __restrict__ y,
                           int C) {
  const int i = blockIdx.x * 256 + threadIdx.x;
  const int total = kB * C * kHW;
  if (i >= total) return;
  const int p = i % kHW;
  const int c = (i / kHW) % C;
  const int b = i / (kHW * C);
  y[i] = t[((size_t)b * kHW + p) * C + c];
}

// diag / anti-diag / col permutation maps (matches _diag_indices)
__global__ void k_build_maps(int* idx, int* inv, int* anti, int* ianti,
                             int* colm) {
  if (blockIdx.x != 0 || threadIdx.x != 0) return;
  int n = 0;
  for (int off = -(kH - 1); off < kW; ++off)
    for (int i = 0; i < kH; ++i) {
      const int j = i + off;
      if (j >= 0 && j < kW) idx[n++] = i * kW + j;
    }
  for (int t = 0; t < kHW; ++t) inv[idx[t]] = t;
  for (int t = 0; t < kHW; ++t) {
    const int p = idx[t];
    anti[t] = (p / kW) * kW + (kW - 1 - (p % kW));
  }
  for (int p = 0; p < kHW; ++p) {
    const int fp = (p / kW) * kW + (kW - 1 - (p % kW));
    ianti[p] = inv[fp];
  }
  for (int p = 0; p < kHW; ++p) colm[p] = (p % kW) * kH + (p / kW);
}

__global__ void k_gather_tok(const float* __restrict__ x,
                             const int* __restrict__ map,
                             float* __restrict__ y, int C) {
  const int i = blockIdx.x * 256 + threadIdx.x;
  const int total = kB * kL * C;
  if (i >= total) return;
  const int c = i % C;
  const int t = (i / C) % kL;
  const int b = i / (C * kL);
  y[i] = x[((size_t)b * kL + map[t]) * C + c];
}

// causal depthwise conv1d (D_CONV=4) + bias + silu; reads x-half of xz (ld 1024)
__global__ void k_conv1d_silu(const float* __restrict__ xz,
                              const float* __restrict__ w,
                              const float* __restrict__ cb,
                              float* __restrict__ out) {
  const int i = blockIdx.x * 256 + threadIdx.x;
  if (i >= kNT * kDI) return;
  const int d = i % kDI;
  const int l = (i / kDI) % kL;
  const int b = i / (kDI * kL);
  float acc = cb[d];
#pragma unroll
  for (int t = 0; t < 4; ++t) {
    const int ls = l - 3 + t;
    if (ls >= 0) acc += xz[((size_t)(b * kL + ls)) * (2 * kDI) + d] * w[d * 4 + t];
  }
  out[(size_t)(b * kL + l) * kDI + d] = acc / (1.f + __expf(-acc));
}

// selective scan: thread = (b,d,s); 16-lane shuffle reduction over s; fused z-gate
__global__ void __launch_bounds__(256) k_scan(
    const float* __restrict__ xc, const float* __restrict__ dt,
    const float* __restrict__ xdbl, const float* __restrict__ A_log,
    const float* __restrict__ Dp, const float* __restrict__ xz,
    float* __restrict__ y) {
  const int tid = blockIdx.x * 256 + threadIdx.x;
  const int s = tid & (kDS - 1);
  const int bd = tid >> 4;
  const int d = bd & (kDI - 1);
  const int b = bd >> 9;  // kDI == 512
  const float Aval = -__expf(A_log[d * kDS + s]);
  const float Dd = Dp[d];
  float h = 0.f;
  for (int l = 0; l < kL; ++l) {
    const size_t tix = (size_t)b * kL + l;
    const float dtv = dt[tix * kDI + d];
    const float xt = xc[tix * kDI + d];
    const float Bv = xdbl[tix * kLdXdbl + kDS + s];
    const float Cv = xdbl[tix * kLdXdbl + 2 * kDS + s];
    h = __expf(dtv * Aval) * h + dtv * xt * Bv;
    float yv = h * Cv;
    yv += __shfl_xor(yv, 1);
    yv += __shfl_xor(yv, 2);
    yv += __shfl_xor(yv, 4);
    yv += __shfl_xor(yv, 8);
    if (s == 0) {
      const float zv = xz[tix * (2 * kDI) + kDI + d];
      y[tix * kDI + d] = (yv + xt * Dd) * (zv / (1.f + __expf(-zv)));
    }
  }
}

__global__ void k_gate(const float* __restrict__ ta, const float* __restrict__ f1,
                       const float* __restrict__ f2, const float* __restrict__ f3,
                       const float* __restrict__ f4, float* __restrict__ out) {
  const int i = blockIdx.x * 256 + threadIdx.x;
  if (i >= kNT * kC) return;
  const float g = 0.25f * (f1[i] + f2[i] + f3[i] + f4[i]);
  out[i] = ta[i] * (1.f / (1.f + __expf(-g)));
}

__global__ void k_dwconv(const float* __restrict__ x, const float* __restrict__ w,
                         const float* __restrict__ bi, float* __restrict__ y,
                         int C, int KH, int KW, int PH, int PW, int act) {
  const int i = blockIdx.x * 256 + threadIdx.x;
  const int total = kB * C * kHW;
  if (i >= total) return;
  const int xx = i % kW;
  const int yy = (i / kW) % kH;
  const int c = (i / kHW) % C;
  const int b = i / (kHW * C);
  float acc = bi[c];
  const float* xp = x + ((size_t)(b * C + c)) * kHW;
  const float* wp = w + (size_t)c * KH * KW;
  for (int ki = 0; ki < KH; ++ki) {
    const int iy = yy + ki - PH;
    if (iy < 0 || iy >= kH) continue;
    for (int kj = 0; kj < KW; ++kj) {
      const int ix = xx + kj - PW;
      if (ix < 0 || ix >= kW) continue;
      acc += xp[iy * kW + ix] * wp[ki * KW + kj];
    }
  }
  y[i] = act_apply(acc, act);
}

__global__ void k_mul(const float* __restrict__ a, const float* __restrict__ b,
                      float* __restrict__ y) {
  const int i = blockIdx.x * 256 + threadIdx.x;
  if (i >= kNT * kC) return;
  y[i] = a[i] * b[i];
}

// out(NCHW) = xin + ls[c] * (tok(token-major) [+ sspat])
__global__ void k_residual(const float* __restrict__ xin,
                           const float* __restrict__ tok,
                           const float* __restrict__ sspat,
                           const float* __restrict__ ls,
                           float* __restrict__ out) {
  const int i = blockIdx.x * 256 + threadIdx.x;
  if (i >= kB * kC * kHW) return;
  const int p = i % kHW;
  const int c = (i / kHW) % kC;
  const int b = i / (kHW * kC);
  float v = tok[((size_t)b * kHW + p) * kC + c];
  if (sspat) v += sspat[i];
  out[i] = xin[i] + ls[c] * v;
}

// ---------------------------------------------------------------- driver
enum {
  IN_X = 0,
  P_BN1_W = 1, P_BN1_B, P_BN2_W, P_BN2_B,
  P_PROJ1_W, P_PROJ1_B, P_PROJ2_W, P_PROJ2_B,
  P_CONV0_W, P_CONV0_B, P_SP1_W, P_SP1_B, P_SP2_W, P_SP2_B,
  P_STRIP_W, P_STRIP_B, P_FC1_W, P_FC1_B, P_FC2_W, P_FC2_B,
  P_DW_W, P_DW_B, P_LS1, P_LS2,
  P_MAMBA_ROW = 25, P_MAMBA_COL = 34, P_MAMBA_DIAG = 43, P_MAMBA_ANTI = 52
};
// mamba param offsets: +0 in_proj_w, +1 conv_w, +2 conv_b, +3 x_proj_w,
// +4 dt_proj_w, +5 dt_proj_b, +6 A_log, +7 D, +8 out_proj_w

extern "C" void kernel_launch(void* const* d_in, const int* in_sizes, int n_in,
                              void* d_out, int out_size, void* d_ws,
                              size_t ws_size, hipStream_t stream) {
  (void)in_sizes; (void)n_in; (void)out_size; (void)ws_size;
  const float* x = (const float*)d_in[IN_X];
  auto P = [&](int i) { return (const float*)d_in[i]; };

  // ---- workspace carve (fp32 arena + maps + f16 weight arena) ----
  float* ws = (float*)d_ws;
  size_t o = 0;
  auto carve = [&](size_t n) { float* p = ws + o; o += n; return p; };
  const size_t NTC = (size_t)kNT * kC;
  const size_t NTDI = (size_t)kNT * kDI;
  const size_t NTXZ = (size_t)kNT * 2 * kDI;
  float* F_S   = carve(NTC);   // bn1 out == residual 's'
  float* F_TH  = carve(NTC);   // gelu(proj1) tokens
  float* F_A   = carve(NTC);   // conv0 out (NCHW)
  float* F_TA  = carve(NTC);   // row tokens of a
  float* F_FR  = carve(NTC);
  float* F_FC  = carve(NTC);
  float* F_FD  = carve(NTC);
  float* F_FA  = carve(NTC);
  float* F_SEQ = carve(NTC);   // mamba input seq / token scratch
  float* F_T1  = carve(NTC);   // token scratch
  float* F_SP1 = carve(NTC);   // spatial scratch
  float* F_SP2 = carve(NTC);   // spatial scratch
  float* F_X1  = carve(NTC);   // x after attn branch
  float* F_XZ  = carve(NTXZ);  // in_proj out / mlp hidden
  float* F_BIG = carve(NTXZ);  // mlp hidden scratch
  float* F_XC  = carve(NTDI);  // conv1d+silu out
  float* F_DT  = carve(NTDI);  // softplus(dt_proj) out
  float* F_Y   = carve(NTDI);  // scan out (z-gated)
  float* F_XD  = carve((size_t)kNT * kLdXdbl);  // x_dbl
  int* maps = (int*)(ws + o);
  int* m_idx = maps;
  int* m_inv = maps + kHW;
  int* m_anti = maps + 2 * kHW;
  int* m_ianti = maps + 3 * kHW;
  int* m_col = maps + 4 * kHW;
  // f16 weight arena (16B aligned: 5*kHW ints = 20480B)
  _Float16* hp = (_Float16*)(maps + 5 * kHW);
  auto carveH = [&](size_t n) { _Float16* p = hp; hp += n; return p; };
  _Float16* hProj1 = carveH((size_t)kC * kC);
  _Float16* hProj2 = carveH((size_t)kC * kC);
  _Float16* hStrip = carveH((size_t)kC * kC);
  _Float16* hFc1   = carveH((size_t)kMlpH * kC);
  _Float16* hFc2   = carveH((size_t)kC * kMlpH);
  _Float16 *hInp[4], *hXp[4], *hDt[4], *hOp[4];
  const int mbase[4] = {P_MAMBA_ROW, P_MAMBA_COL, P_MAMBA_DIAG, P_MAMBA_ANTI};
  for (int m = 0; m < 4; ++m) {
    hInp[m] = carveH((size_t)2 * kDI * kC);
    hXp[m]  = carveH((size_t)kLdXdbl * kDI);
    hDt[m]  = carveH((size_t)kDI * kDS);
    hOp[m]  = carveH((size_t)kC * kDI);
  }

  auto ew = [](size_t n) { return dim3((unsigned)((n + 255) / 256)); };
  auto cvt = [&](const float* s, _Float16* d, size_t n) {
    k_cvt_h<<<ew(n), 256, 0, stream>>>(s, d, (int)n);
  };
  auto gemm = [&](const float* Ap, int lda, const _Float16* Wp, int ldw,
                  const float* bp, float* Yp, int ldy, int M, int N, int K,
                  int act) {
    dim3 g((unsigned)((N + 63) / 64), (unsigned)((M + 63) / 64));
    k_gemm_wmma<<<g, 256, 0, stream>>>(Ap, lda, Wp, ldw, bp, Yp, ldy, M, N, K, act);
  };
  auto run_mamba = [&](const float* seq, int pb, int mi, float* feat_out) {
    gemm(seq, kC, hInp[mi], kC, nullptr, F_XZ, 2 * kDI, kNT, 2 * kDI, kC, ACT_NONE);
    k_conv1d_silu<<<ew(NTDI), 256, 0, stream>>>(F_XZ, P(pb + 1), P(pb + 2), F_XC);
    gemm(F_XC, kDI, hXp[mi], kDI, nullptr, F_XD, kLdXdbl, kNT, kLdXdbl, kDI, ACT_NONE);
    gemm(F_XD, kLdXdbl, hDt[mi], kDS, P(pb + 5), F_DT, kDI, kNT, kDI, kDS, ACT_SOFTPLUS);
    k_scan<<<dim3((kB * kDI * kDS) / 256), 256, 0, stream>>>(
        F_XC, F_DT, F_XD, P(pb + 6), P(pb + 7), F_XZ, F_Y);
    gemm(F_Y, kDI, hOp[mi], kDI, nullptr, feat_out, kC, kNT, kC, kDI, ACT_NONE);
  };

  // ---- one-time weight conversions + permutation maps ----
  k_build_maps<<<1, 1, 0, stream>>>(m_idx, m_inv, m_anti, m_ianti, m_col);
  cvt(P(P_PROJ1_W), hProj1, (size_t)kC * kC);
  cvt(P(P_PROJ2_W), hProj2, (size_t)kC * kC);
  cvt(P(P_STRIP_W), hStrip, (size_t)kC * kC);
  cvt(P(P_FC1_W), hFc1, (size_t)kMlpH * kC);
  cvt(P(P_FC2_W), hFc2, (size_t)kC * kMlpH);
  for (int m = 0; m < 4; ++m) {
    cvt(P(mbase[m] + 0), hInp[m], (size_t)2 * kDI * kC);
    cvt(P(mbase[m] + 3), hXp[m], (size_t)kLdXdbl * kDI);
    cvt(P(mbase[m] + 4), hDt[m], (size_t)kDI * kDS);
    cvt(P(mbase[m] + 8), hOp[m], (size_t)kC * kDI);
  }

  // ---- attn branch ----
  k_bn<<<kC, 256, 0, stream>>>(x, P(P_BN1_W), P(P_BN1_B), F_S);
  k_nchw2tok<<<ew(NTC), 256, 0, stream>>>(F_S, F_T1, kC, 0);
  gemm(F_T1, kC, hProj1, kC, P(P_PROJ1_B), F_TH, kC, kNT, kC, kC, ACT_GELU);
  k_tok2nchw<<<ew(NTC), 256, 0, stream>>>(F_TH, F_SP1, kC);
  k_dwconv<<<ew(NTC), 256, 0, stream>>>(F_SP1, P(P_CONV0_W), P(P_CONV0_B), F_A,
                                        kC, 5, 5, 2, 2, ACT_NONE);
  k_nchw2tok<<<ew(NTC), 256, 0, stream>>>(F_A, F_TA, kC, 0);

  // row
  run_mamba(F_TA, P_MAMBA_ROW, 0, F_FR);
  // col
  k_nchw2tok<<<ew(NTC), 256, 0, stream>>>(F_A, F_SEQ, kC, 1);
  run_mamba(F_SEQ, P_MAMBA_COL, 1, F_T1);
  k_gather_tok<<<ew(NTC), 256, 0, stream>>>(F_T1, m_col, F_FC, kC);
  // diag
  k_gather_tok<<<ew(NTC), 256, 0, stream>>>(F_TA, m_idx, F_SEQ, kC);
  run_mamba(F_SEQ, P_MAMBA_DIAG, 2, F_T1);
  k_gather_tok<<<ew(NTC), 256, 0, stream>>>(F_T1, m_inv, F_FD, kC);
  // anti
  k_gather_tok<<<ew(NTC), 256, 0, stream>>>(F_TA, m_anti, F_SEQ, kC);
  run_mamba(F_SEQ, P_MAMBA_ANTI, 3, F_T1);
  k_gather_tok<<<ew(NTC), 256, 0, stream>>>(F_T1, m_ianti, F_FA, kC);

  // gate + strip convs + strip1x1
  k_gate<<<ew(NTC), 256, 0, stream>>>(F_TA, F_FR, F_FC, F_FD, F_FA, F_T1);
  k_tok2nchw<<<ew(NTC), 256, 0, stream>>>(F_T1, F_SP1, kC);
  k_dwconv<<<ew(NTC), 256, 0, stream>>>(F_SP1, P(P_SP1_W), P(P_SP1_B), F_SP2,
                                        kC, 1, 19, 0, 9, ACT_NONE);
  k_dwconv<<<ew(NTC), 256, 0, stream>>>(F_SP2, P(P_SP2_W), P(P_SP2_B), F_SP1,
                                        kC, 19, 1, 9, 0, ACT_NONE);
  k_nchw2tok<<<ew(NTC), 256, 0, stream>>>(F_SP1, F_T1, kC, 0);
  gemm(F_T1, kC, hStrip, kC, P(P_STRIP_B), F_SEQ, kC, kNT, kC, kC, ACT_NONE);
  k_mul<<<ew(NTC), 256, 0, stream>>>(F_TH, F_SEQ, F_T1);
  gemm(F_T1, kC, hProj2, kC, P(P_PROJ2_B), F_SEQ, kC, kNT, kC, kC, ACT_NONE);
  k_residual<<<ew(NTC), 256, 0, stream>>>(x, F_SEQ, F_S, P(P_LS1), F_X1);

  // ---- mlp branch ----
  k_bn<<<kC, 256, 0, stream>>>(F_X1, P(P_BN2_W), P(P_BN2_B), F_SP1);
  k_nchw2tok<<<ew(NTC), 256, 0, stream>>>(F_SP1, F_T1, kC, 0);
  gemm(F_T1, kC, hFc1, kC, P(P_FC1_B), F_XZ, kMlpH, kNT, kMlpH, kC, ACT_NONE);
  k_tok2nchw<<<ew((size_t)kNT * kMlpH), 256, 0, stream>>>(F_XZ, F_BIG, kMlpH);
  k_dwconv<<<ew((size_t)kNT * kMlpH), 256, 0, stream>>>(
      F_BIG, P(P_DW_W), P(P_DW_B), F_XZ, kMlpH, 3, 3, 1, 1, ACT_GELU);
  k_nchw2tok<<<ew((size_t)kNT * kMlpH), 256, 0, stream>>>(F_XZ, F_BIG, kMlpH, 0);
  gemm(F_BIG, kMlpH, hFc2, kMlpH, P(P_FC2_B), F_T1, kC, kNT, kC, kMlpH, ACT_NONE);
  k_residual<<<ew(NTC), 256, 0, stream>>>(F_X1, F_T1, nullptr, P(P_LS2),
                                          (float*)d_out);
}
